// MambaBlock_55551107006692
// MI455X (gfx1250) — matrix operand order repaired and checked by reading, not compiled, not run on hardware
//
#include <hip/hip_runtime.h>
#include <hip/hip_bf16.h>

// ---------------------------------------------------------------------------
// Mamba block forward for MI455X (gfx1250):
//   bf16 WMMA GEMMs with double-buffered async-to-LDS B-panel staging
//   + fused selective scan.
// ---------------------------------------------------------------------------

#define D_MODEL 1024
#define D_STATE 16
#define D_CONV  4
#define D_INNER 2048
#define DT_RANK 128
#define N_PROJ  (DT_RANK + 2 * D_STATE)   // 160
#define BATCH   2
#define SEQ     1024
#define MROWS   (BATCH * SEQ)             // 2048

typedef __attribute__((ext_vector_type(16))) __bf16 v16bf;
typedef __attribute__((ext_vector_type(8)))  __bf16 v8bf;
typedef __attribute__((ext_vector_type(8)))  float  v8f;

// round-to-nearest-even f32 -> bf16
static __device__ __forceinline__ __bf16 f2bf(float f) {
    unsigned u = __builtin_bit_cast(unsigned, f);
    unsigned r = (u + 0x7FFFu + ((u >> 16) & 1u)) >> 16;
    unsigned short s = (unsigned short)r;
    return __builtin_bit_cast(__bf16, s);
}

// A-fragment (16x32 MxK, 16-bit): lane holds row (lane&15);
// halves 0..7 = K[koff .. +7], halves 8..15 = K[koff+8 .. +15],
// koff = (lane>>4)*16.  Two aligned 16-byte global loads.
static __device__ __forceinline__ v16bf load_frag_a(const __bf16* row, int k0, int koff) {
    v8bf lo = *reinterpret_cast<const v8bf*>(row + k0 + koff);
    v8bf hi = *reinterpret_cast<const v8bf*>(row + k0 + koff + 8);
    return __builtin_shufflevector(lo, hi, 0,1,2,3,4,5,6,7,8,9,10,11,12,13,14,15);
}

// issue one 16-byte async global->LDS copy (tracked by ASYNCcnt)
static __device__ __forceinline__ void async_copy16(unsigned ldsDst, const __bf16* src) {
    asm volatile("global_load_async_to_lds_b128 %0, %1, off"
                 :: "v"(ldsDst),
                    "v"((unsigned long long)(uintptr_t)src)
                 : "memory");
}

// ---------------------------------------------------------------------------
// cast f32 -> bf16
// ---------------------------------------------------------------------------
__global__ void cast_bf16_kernel(__bf16* __restrict__ dst,
                                 const float* __restrict__ src, int n) {
    int i = blockIdx.x * 256 + threadIdx.x;
    if (i < n) dst[i] = f2bf(src[i]);
}

// ---------------------------------------------------------------------------
// C(MxN,f32) = A(MxK,bf16,row,lda) @ B(NxK,bf16,row,ldb)^T
//
// Block (8 waves) owns a 128(M) x 64(N) tile. Double-buffered pipeline:
//   iter i: issue async copy of B panel i+1 into buf[(i+1)&1]
//           s_wait_asynccnt 1  (async loads complete in order -> panel i ready)
//           barrier; A frag from global + 4 B frags via ds_load_b128
//           4x v_wmma_f32_16x16x32_bf16; barrier
// The 4 KB panel copy for step i+1 overlaps the WMMAs of step i, and all
// 8 waves share one panel (8x less B traffic than per-wave global fetch).
// mode 0: plain store.  mode 1: softplus(acc + bias[n]).
// Requires M % 128 == 0, K % 32 == 0, N % 16 == 0 (true at all call sites).
// ---------------------------------------------------------------------------
__global__ __launch_bounds__(256)
void gemm_bf16_wmma_kernel(float* __restrict__ C,
                           const __bf16* __restrict__ A,
                           const __bf16* __restrict__ B,
                           int M, int N, int K, int lda, int ldb, int ldc,
                           const float* __restrict__ bias, int mode) {
    __shared__ __align__(32) __bf16 sB[2][64 * 32];   // 2 x 4 KB B panels

    const int tid      = threadIdx.x;
    const int lane     = tid & 31;
    const int waveId   = tid >> 5;
    const int nStrips  = (N + 63) >> 6;
    const int bm       = blockIdx.x / nStrips;
    const int strip    = blockIdx.x - bm * nStrips;
    const int m0       = (bm * 8 + waveId) << 4;
    const int n0       = strip << 6;
    const int mr       = lane & 15;
    const int halfSel  = lane >> 4;
    const int koffA    = halfSel << 4;    // {0, 16}
    const int koffB    = halfSel << 4;    // {0, 16}

    const __bf16* aRow = A + (size_t)(m0 + mr) * lda;

    // async-copy mapping: thread tid moves 16 bytes per panel:
    //   B[n0 + (tid>>2)][k0 + (tid&3)*8 .. +7] -> sB[buf][tid>>2][(tid&3)*8 ..]
    int scol = n0 + (tid >> 2);
    if (scol >= N) scol = 0;                          // safe row, never consumed
    const __bf16* srcB = B + (size_t)scol * ldb + (tid & 3) * 8;
    const unsigned ldsBase = (unsigned)(uintptr_t)(void*)&sB[0][0];
    const unsigned ldsDst0 = ldsBase + (unsigned)tid * 16u;
    const unsigned ldsDst1 = ldsDst0 + 64u * 32u * 2u;

    bool valid[4];
#pragma unroll
    for (int t = 0; t < 4; ++t) valid[t] = (n0 + t * 16) < N;   // N mult of 16

    v8f acc0 = {}, acc1 = {}, acc2 = {}, acc3 = {};

    const int steps = K >> 5;
    async_copy16(ldsDst0, srcB);                      // prologue: panel 0
    for (int i = 0; i < steps; ++i) {
        const int k0 = i << 5;
        const int cur = i & 1;
        if (i + 1 < steps) {
            // overlap: stream panel i+1 while computing panel i
            async_copy16(cur ? ldsDst0 : ldsDst1, srcB + k0 + 32);
            asm volatile("s_wait_asynccnt 0x1" ::: "memory");  // panel i resident
        } else {
            asm volatile("s_wait_asynccnt 0x0" ::: "memory");
        }
        __syncthreads();

        v16bf a  = load_frag_a(aRow, k0, koffA);
        const __bf16* sb = &sB[cur][0];
        v16bf b0 = *reinterpret_cast<const v16bf*>(sb + (0 * 16 + mr) * 32 + koffB);
        v16bf b1 = *reinterpret_cast<const v16bf*>(sb + (1 * 16 + mr) * 32 + koffB);
        v16bf b2 = *reinterpret_cast<const v16bf*>(sb + (2 * 16 + mr) * 32 + koffB);
        v16bf b3 = *reinterpret_cast<const v16bf*>(sb + (3 * 16 + mr) * 32 + koffB);

        acc0 = __builtin_amdgcn_wmma_f32_16x16x32_bf16(false, a, false, b0, (short)0, acc0, false, false);
        acc1 = __builtin_amdgcn_wmma_f32_16x16x32_bf16(false, a, false, b1, (short)0, acc1, false, false);
        acc2 = __builtin_amdgcn_wmma_f32_16x16x32_bf16(false, a, false, b2, (short)0, acc2, false, false);
        acc3 = __builtin_amdgcn_wmma_f32_16x16x32_bf16(false, a, false, b3, (short)0, acc3, false, false);

        __syncthreads();   // all waves done reading buf[cur] -> reusable
    }

    v8f accs[4] = {acc0, acc1, acc2, acc3};
#pragma unroll
    for (int t = 0; t < 4; ++t) {
        if (!valid[t]) continue;
        const int n    = n0 + t * 16 + mr;
        const float bv = (mode == 1) ? bias[n] : 0.f;
#pragma unroll
        for (int r = 0; r < 8; ++r) {
            const int m = m0 + r + (halfSel << 3);
            float v = accs[t][r];
            if (mode == 1) {
                v += bv;
                v = (v > 20.f) ? v : log1pf(expf(v));   // softplus
            }
            C[(size_t)m * ldc + n] = v;
        }
    }
}

// ---------------------------------------------------------------------------
// depthwise causal conv1d(k=4) + bias + SiLU; reads xc = xz[..., :D_INNER]
// ---------------------------------------------------------------------------
__global__ __launch_bounds__(256)
void conv_silu_kernel(float* __restrict__ u, __bf16* __restrict__ ubf,
                      const float* __restrict__ xz,
                      const float* __restrict__ cw,
                      const float* __restrict__ cb) {
    int idx = blockIdx.x * 256 + threadIdx.x;
    if (idx >= MROWS * D_INNER) return;
    int row = idx >> 11;          // b*SEQ + l
    int d   = idx & (D_INNER - 1);
    int l   = row & (SEQ - 1);
    float acc = cb[d];
#pragma unroll
    for (int j = 0; j < D_CONV; ++j) {
        int ll = l - (D_CONV - 1) + j;
        if (ll >= 0)
            acc += xz[(size_t)(row - (D_CONV - 1) + j) * (2 * D_INNER) + d] * cw[d * D_CONV + j];
    }
    float s = acc / (1.f + expf(-acc));   // silu
    u[idx]   = s;
    ubf[idx] = f2bf(s);
}

// ---------------------------------------------------------------------------
// fused selective scan + u*D + silu(z) gating -> bf16 (GEMM-ready).
// One thread owns one (b, d) chain; 16-entry state lives in VGPRs.
// ---------------------------------------------------------------------------
__global__ __launch_bounds__(256)
void scan_gate_kernel(__bf16* __restrict__ ybf,
                      const float* __restrict__ dt,
                      const float* __restrict__ u,
                      const float* __restrict__ xdbl,   // (MROWS, 160)
                      const float* __restrict__ xz,     // z = cols [D_INNER, 2*D_INNER)
                      const float* __restrict__ A_log,
                      const float* __restrict__ Dv) {
    int g = blockIdx.x * 256 + threadIdx.x;
    if (g >= BATCH * D_INNER) return;
    int b = g >> 11;
    int d = g & (D_INNER - 1);

    float An[D_STATE], h[D_STATE];
#pragma unroll
    for (int n = 0; n < D_STATE; ++n) {
        An[n] = -expf(A_log[d * D_STATE + n]);
        h[n]  = 0.f;
    }
    const float Dd = Dv[d];

    for (int l = 0; l < SEQ; ++l) {
        const int row   = (b << 10) + l;
        const float dtv = dt[(size_t)row * D_INNER + d];
        const float uv  = u [(size_t)row * D_INNER + d];
        const float zv  = xz[(size_t)row * (2 * D_INNER) + D_INNER + d];
        const float du  = dtv * uv;
        const float* bc = xdbl + (size_t)row * N_PROJ + DT_RANK;  // B then C
        float y = 0.f;
#pragma unroll
        for (int n = 0; n < D_STATE; ++n) {
            float dA = expf(dtv * An[n]);
            h[n] = dA * h[n] + du * bc[n];
            y += h[n] * bc[D_STATE + n];
        }
        y += uv * Dd;
        const float sz = zv / (1.f + expf(-zv));
        ybf[(size_t)row * D_INNER + d] = f2bf(y * sz);
    }
}

// ---------------------------------------------------------------------------
// host-side orchestration
// ---------------------------------------------------------------------------
static inline size_t align256(size_t x) { return (x + 255) & ~(size_t)255; }

static inline void launch_gemm(float* C, const __bf16* A, const __bf16* B,
                               int M, int N, int K, int lda, int ldb, int ldc,
                               const float* bias, int mode, hipStream_t s) {
    int nStrips = (N + 63) >> 6;
    int blocks  = (M >> 7) * nStrips;   // M % 128 == 0 at every call site
    gemm_bf16_wmma_kernel<<<blocks, 256, 0, s>>>(C, A, B, M, N, K, lda, ldb, ldc, bias, mode);
}

extern "C" void kernel_launch(void* const* d_in, const int* in_sizes, int n_in,
                              void* d_out, int out_size, void* d_ws, size_t ws_size,
                              hipStream_t stream) {
    const float* x      = (const float*)d_in[0];   // (2,1024,1024)
    const float* W_in   = (const float*)d_in[1];   // (4096,1024)
    const float* conv_w = (const float*)d_in[2];   // (2048,1,4)
    const float* conv_b = (const float*)d_in[3];   // (2048,)
    const float* W_x    = (const float*)d_in[4];   // (160,2048)
    const float* W_dt   = (const float*)d_in[5];   // (2048,128)
    const float* b_dt   = (const float*)d_in[6];   // (2048,)
    const float* A_log  = (const float*)d_in[7];   // (2048,16)
    const float* Dvec   = (const float*)d_in[8];   // (2048,)
    const float* W_out  = (const float*)d_in[9];   // (1024,2048)
    float* out = (float*)d_out;                    // (2,1024,1024)

    // workspace carve-up
    char* base = (char*)d_ws;
    size_t cur = 0;
    auto take = [&](size_t bytes) -> char* {
        char* p = base + cur;
        cur = align256(cur + bytes);
        return p;
    };
    float*  xz      = (float*) take((size_t)MROWS * 2 * D_INNER * 4);   // 32 MB
    __bf16* x_bf    = (__bf16*)take((size_t)MROWS * D_MODEL * 2);
    __bf16* win_bf  = (__bf16*)take((size_t)2 * D_INNER * D_MODEL * 2);
    float*  u_f     = (float*) take((size_t)MROWS * D_INNER * 4);
    __bf16* u_bf    = (__bf16*)take((size_t)MROWS * D_INNER * 2);
    __bf16* wx_bf   = (__bf16*)take((size_t)N_PROJ * D_INNER * 2);
    float*  xdbl    = (float*) take((size_t)MROWS * N_PROJ * 4);
    __bf16* xdbl_bf = (__bf16*)take((size_t)MROWS * N_PROJ * 2);
    __bf16* wdt_bf  = (__bf16*)take((size_t)D_INNER * DT_RANK * 2);
    float*  dt_f    = (float*) take((size_t)MROWS * D_INNER * 4);
    __bf16* y_bf    = (__bf16*)take((size_t)MROWS * D_INNER * 2);
    __bf16* wout_bf = (__bf16*)take((size_t)D_MODEL * D_INNER * 2);
    (void)ws_size;

    auto cast = [&](__bf16* dst, const float* src, int n) {
        cast_bf16_kernel<<<(n + 255) / 256, 256, 0, stream>>>(dst, src, n);
    };

    // 0) bf16 casts of activations/weights (deterministic each call)
    cast(x_bf,    x,     MROWS * D_MODEL);
    cast(win_bf,  W_in,  2 * D_INNER * D_MODEL);
    cast(wx_bf,   W_x,   N_PROJ * D_INNER);
    cast(wdt_bf,  W_dt,  D_INNER * DT_RANK);
    cast(wout_bf, W_out, D_MODEL * D_INNER);

    // 1) xz = x @ W_in^T            (2048 x 4096 x 1024)
    launch_gemm(xz, x_bf, win_bf, MROWS, 2 * D_INNER, D_MODEL,
                D_MODEL, D_MODEL, 2 * D_INNER, nullptr, 0, stream);

    // 2) depthwise conv + SiLU -> u (f32 + bf16)
    conv_silu_kernel<<<(MROWS * D_INNER + 255) / 256, 256, 0, stream>>>(
        u_f, u_bf, xz, conv_w, conv_b);

    // 3) x_dbl = u @ W_x^T          (2048 x 160 x 2048)
    launch_gemm(xdbl, u_bf, wx_bf, MROWS, N_PROJ, D_INNER,
                D_INNER, D_INNER, N_PROJ, nullptr, 0, stream);
    cast(xdbl_bf, xdbl, MROWS * N_PROJ);

    // 4) dt = softplus(x_dbl[:, :128] @ W_dt^T + b_dt)   (2048 x 2048 x 128)
    launch_gemm(dt_f, xdbl_bf, wdt_bf, MROWS, D_INNER, DT_RANK,
                N_PROJ, DT_RANK, D_INNER, b_dt, 1, stream);

    // 5) fused selective scan + u*D + silu(z) gating -> y_bf
    scan_gate_kernel<<<(BATCH * D_INNER + 255) / 256, 256, 0, stream>>>(
        y_bf, dt_f, u_f, xdbl, xz, A_log, Dvec);

    // 6) out = y @ W_out^T          (2048 x 1024 x 2048)
    launch_gemm(out, y_bf, wout_bf, MROWS, D_MODEL, D_INNER,
                D_INNER, D_INNER, D_MODEL, nullptr, 0, stream);
}
